// SimpleGCNClassifier_33990371181249
// MI455X (gfx1250) — matrix operand order repaired and checked by reading, not compiled, not run on hardware
//
#include <hip/hip_runtime.h>
#include <hip/hip_bf16.h>

#define N_NODES   100000
#define N_EDGES   600000
#define IN_DIM    128
#define HID       128
#define N_GRAPHS  512
#define N_CLASSES 10

typedef __attribute__((ext_vector_type(16))) _Float16 v16h;
typedef __attribute__((ext_vector_type(8)))  float    v8f;

// ---------------- workspace layout (float offsets) ----------------
#define WS_AGG   0            // N_NODES*128 = 12,800,000 floats
#define WS_NSRC  12800000     // 100,000 (degree then rsqrt-norm, in place)
#define WS_NDST  12900000     // 100,000
#define WS_GSUM  13000000     // 512*128 = 65,536
#define WS_CNT   13065536     // 512
#define WS_WH    13066048     // 16,384 _Float16 (8,192 floats)
#define WS_ZERO_F4 3266512    // 13,066,048 / 4

// ---------------- zero accumulators (float4 stores) ----------------
__global__ void k_zero4(float4* __restrict__ p, int n4) {
    int i = blockIdx.x * 256 + threadIdx.x;
    if (i < n4) p[i] = make_float4(0.f, 0.f, 0.f, 0.f);
}

// ---------------- degree counting over edges ----------------
__global__ void k_deg(const int* __restrict__ src, const int* __restrict__ dst,
                      float* __restrict__ deg_out, float* __restrict__ deg_in) {
    int e = blockIdx.x * 256 + threadIdx.x;
    if (e < N_EDGES) {
        atomicAdd(&deg_out[src[e]], 1.0f);
        atomicAdd(&deg_in[dst[e]], 1.0f);
    }
}

// ---------------- deg -> norm (in place), plus per-graph node counts ----------------
__global__ void k_norm_cnt(float* __restrict__ nsrc, float* __restrict__ ndst,
                           const int* __restrict__ gids, float* __restrict__ cnt) {
    int i = blockIdx.x * 256 + threadIdx.x;
    if (i < N_NODES) {
        nsrc[i] = rsqrtf(fmaxf(nsrc[i], 1.0f));
        ndst[i] = rsqrtf(fmaxf(ndst[i], 1.0f));
        atomicAdd(&cnt[gids[i]], 1.0f);
    }
}

// ---------------- W [K][N] f32 -> Wh [N][K] f16 (column-major for B frags) ----------
__global__ void k_wt(const float* __restrict__ W, _Float16* __restrict__ Wh) {
    int t = blockIdx.x * 256 + threadIdx.x;   // t < 128*128
    int n = t >> 7, k = t & 127;
    Wh[(size_t)n * 128 + k] = (_Float16)W[(size_t)k * 128 + n];
}

// ---------------- edge scatter: agg[dst] += feats[src] * norm_src[src] -----------
// 32 lanes per edge, float4 each -> coalesced 512B feature read per edge.
__global__ void k_scatter(const float* __restrict__ feats, const float* __restrict__ nsrc,
                          const int* __restrict__ src, const int* __restrict__ dst,
                          float* __restrict__ agg) {
    long long gid = (long long)blockIdx.x * 256 + threadIdx.x;
    int e  = (int)(gid >> 5);
    int c4 = ((int)gid & 31) * 4;
    if (e >= N_EDGES) return;
    int s = src[e], d = dst[e];
    float ns = nsrc[s];
    float4 f = *(const float4*)(feats + (size_t)s * 128 + c4);
    float* ap = agg + (size_t)d * 128 + c4;
    atomicAdd(ap + 0, f.x * ns);
    atomicAdd(ap + 1, f.y * ns);
    atomicAdd(ap + 2, f.z * ns);
    atomicAdd(ap + 3, f.w * ns);
}

// ---------------- fused GEMM + ReLU + graph-pool ----------------
// H = relu((agg * norm_dst) @ W + b); gsum[g] += H[node]  (atomic pool)
// One wave per 16-row strip; 4x v_wmma_f32_16x16x32_f16 per 16x16 output tile.
__global__ void __launch_bounds__(256)
k_gemm_pool(const float* __restrict__ agg, const float* __restrict__ ndst,
            const _Float16* __restrict__ Wh, const float* __restrict__ bias,
            const int* __restrict__ gids, float* __restrict__ gsum) {
    const int lane = threadIdx.x & 31;
    const int wave = threadIdx.x >> 5;
    const int gw   = blockIdx.x * 8 + wave;        // global wave index
    const int node_base = gw * 16;
    if (node_base >= N_NODES) return;              // wave-uniform: EXEC all-1s inside

    const int m16  = lane & 15;
    const int ksel = lane >> 4;

    // ---- A fragments: 16x32 f16 per chunk, rows = node_base..+15, scaled by norm_dst
    // lanes 0-15: elems 0..15 -> K=kOff+0..15
    // lanes16-31: elems 0..15 -> K=kOff+16..31
    const int nodeA = node_base + m16;
    const float nd  = ndst[nodeA];
    const float4* arow = (const float4*)(agg + (size_t)nodeA * 128);

    v16h a[4];
#pragma unroll
    for (int kc = 0; kc < 4; ++kc) {
        int q0 = (kc * 32 + ksel * 16) >> 2;       // float4 index of K-run 1
        int q1 = (kc * 32 + 8 + ksel * 16) >> 2;   // float4 index of K-run 2
        float4 f0 = arow[q0],  f1 = arow[q0 + 1];
        float4 f2 = arow[q1],  f3 = arow[q1 + 1];
        a[kc][0]  = (_Float16)(f0.x * nd); a[kc][1]  = (_Float16)(f0.y * nd);
        a[kc][2]  = (_Float16)(f0.z * nd); a[kc][3]  = (_Float16)(f0.w * nd);
        a[kc][4]  = (_Float16)(f1.x * nd); a[kc][5]  = (_Float16)(f1.y * nd);
        a[kc][6]  = (_Float16)(f1.z * nd); a[kc][7]  = (_Float16)(f1.w * nd);
        a[kc][8]  = (_Float16)(f2.x * nd); a[kc][9]  = (_Float16)(f2.y * nd);
        a[kc][10] = (_Float16)(f2.z * nd); a[kc][11] = (_Float16)(f2.w * nd);
        a[kc][12] = (_Float16)(f3.x * nd); a[kc][13] = (_Float16)(f3.y * nd);
        a[kc][14] = (_Float16)(f3.z * nd); a[kc][15] = (_Float16)(f3.w * nd);
    }

    // graph ids for the 8 D rows this lane owns (M = r + 8*ksel)
    int gid8[8];
#pragma unroll
    for (int r = 0; r < 8; ++r) gid8[r] = gids[node_base + ksel * 8 + r];

    const int kb = ksel * 16;                      // B: lanes 0-15 K=0..15, 16-31 K=16..31
#pragma unroll
    for (int nt = 0; nt < 8; ++nt) {
        const int coln = nt * 16 + m16;            // D/B column owned by this lane
        const _Float16* wcol = Wh + (size_t)coln * 128 + kb;
        v16h b0 = *(const v16h*)(wcol);
        v16h b1 = *(const v16h*)(wcol + 32);
        v16h b2 = *(const v16h*)(wcol + 64);
        v16h b3 = *(const v16h*)(wcol + 96);

        float bv = bias[coln];
        v8f c = {bv, bv, bv, bv, bv, bv, bv, bv};
        c = __builtin_amdgcn_wmma_f32_16x16x32_f16(false, a[0], false, b0, (short)0, c, false, false);
        c = __builtin_amdgcn_wmma_f32_16x16x32_f16(false, a[1], false, b1, (short)0, c, false, false);
        c = __builtin_amdgcn_wmma_f32_16x16x32_f16(false, a[2], false, b2, (short)0, c, false, false);
        c = __builtin_amdgcn_wmma_f32_16x16x32_f16(false, a[3], false, b3, (short)0, c, false, false);

#pragma unroll
        for (int r = 0; r < 8; ++r) {
            float v = c[r] > 0.0f ? c[r] : 0.0f;   // ReLU
            atomicAdd(&gsum[(size_t)gid8[r] * 128 + coln], v);
        }
    }
}

// ---------------- mean + 2-layer MLP head ----------------
__global__ void k_head(const float* __restrict__ gsum, const float* __restrict__ cnt,
                       const float* __restrict__ W1, const float* __restrict__ b1,
                       const float* __restrict__ W2, const float* __restrict__ b2,
                       float* __restrict__ out) {
    int g = blockIdx.x * 256 + threadIdx.x;
    if (g >= N_GRAPHS) return;
    float c = cnt[g];
    float inv = (c > 0.0f) ? (1.0f / fmaxf(c, 1.0f)) : 0.0f;
    const float* row = gsum + (size_t)g * 128;
    float t[12];
#pragma unroll
    for (int j = 0; j < 12; ++j) t[j] = b1[j];
    for (int k = 0; k < 128; ++k) {
        float hk = row[k] * inv;
#pragma unroll
        for (int j = 0; j < 12; ++j) t[j] += hk * W1[k * 12 + j];
    }
#pragma unroll
    for (int o = 0; o < N_CLASSES; ++o) {
        float s = b2[o];
#pragma unroll
        for (int j = 0; j < 12; ++j) s += t[j] * W2[j * 10 + o];
        out[(size_t)g * N_CLASSES + o] = s;
    }
}

extern "C" void kernel_launch(void* const* d_in, const int* in_sizes, int n_in,
                              void* d_out, int out_size, void* d_ws, size_t ws_size,
                              hipStream_t stream) {
    const float* feats = (const float*)d_in[0];
    const float* W     = (const float*)d_in[1];
    const float* b     = (const float*)d_in[2];
    const float* W1    = (const float*)d_in[3];
    const float* b1    = (const float*)d_in[4];
    const float* W2    = (const float*)d_in[5];
    const float* b2    = (const float*)d_in[6];
    const int*   src   = (const int*)d_in[7];
    const int*   dst   = (const int*)d_in[8];
    const int*   gids  = (const int*)d_in[9];

    float* ws   = (float*)d_ws;
    float* agg  = ws + WS_AGG;
    float* nsrc = ws + WS_NSRC;
    float* ndst = ws + WS_NDST;
    float* gsum = ws + WS_GSUM;
    float* cnt  = ws + WS_CNT;
    _Float16* Wh = (_Float16*)(ws + WS_WH);
    float* out = (float*)d_out;

    // 1) zero accumulators (agg, degs, gsum, cnt)
    k_zero4<<<(WS_ZERO_F4 + 255) / 256, 256, 0, stream>>>((float4*)ws, WS_ZERO_F4);
    // 2) degrees
    k_deg<<<(N_EDGES + 255) / 256, 256, 0, stream>>>(src, dst, nsrc, ndst);
    // 3) norms + per-graph node counts
    k_norm_cnt<<<(N_NODES + 255) / 256, 256, 0, stream>>>(nsrc, ndst, gids, cnt);
    // 4) W -> f16 transposed
    k_wt<<<(HID * HID) / 256, 256, 0, stream>>>(W, Wh);
    // 5) edge gather-scatter SpMM
    k_scatter<<<(N_EDGES * 32) / 256, 256, 0, stream>>>(feats, nsrc, src, dst, agg);
    // 6) fused WMMA GEMM + ReLU + pooled segment-sum  (6250 waves, 8 waves/block)
    k_gemm_pool<<<(6250 + 7) / 8, 256, 0, stream>>>(agg, ndst, Wh, b, gids, gsum);
    // 7) mean + MLP head
    k_head<<<(N_GRAPHS + 255) / 256, 256, 0, stream>>>(gsum, cnt, W1, b1, W2, b2, out);
}
